// BiDAF_49821620633734
// MI455X (gfx1250) — matrix operand order repaired and checked by reading, not compiled, not run on hardware
//
#include <hip/hip_runtime.h>

// =====================================================================
// BiDAF forward for MI455X (gfx1250, wave32, WMMA).
// Dense projections: v_wmma_f32_16x16x32_f16 (f16 in, f32 accumulate).
// LSTM scans: persistent one-WGP kernels; h + per-step gate-bias tile in
// LDS; xw[t+1] tile streamed in with the Tensor Data Mover (TDM) while
// step t computes (double-buffered), completion via s_wait_tensorcnt.
// =====================================================================

typedef __attribute__((ext_vector_type(16))) _Float16 v16h;
typedef __attribute__((ext_vector_type(8)))  float    v8f;
typedef __attribute__((ext_vector_type(4)))  unsigned int v4u;
typedef __attribute__((ext_vector_type(8)))  int      v8i;
typedef __attribute__((ext_vector_type(4)))  int      v4i;

union HF16 { v16h v; _Float16 h[16]; };

#define BB   64
#define TCN  512
#define TQW  64
#define NG   448   // 4 gates x 112 (100 padded to 112 = 7 WMMA tiles)
#define KHP  128   // hidden 100 padded to 128 for WMMA K
#define SCANT 896  // 28 waves: 4 m-tiles x 7 n-groups (4 n-tiles each)

#if defined(__has_builtin)
#if __has_builtin(__builtin_amdgcn_tensor_load_to_lds) && __has_builtin(__builtin_amdgcn_s_wait_tensorcnt)
#define HAVE_TDM 1
#endif
#endif

// ---------------------------------------------------------------------
// device helpers
// ---------------------------------------------------------------------
__device__ __forceinline__ float sigf(float x) { return 1.0f / (1.0f + __expf(-x)); }

// A fragment (16x32 f16, row-major MxK source). Contiguous k order (same as B frags):
// lane&15 = M row; lanes 0-15 carry K 0-15, lanes 16-31 carry K 16-31.
__device__ __forceinline__ v16h frag_a_f16(const _Float16* __restrict__ A,
                                           int row0, int lda, int k0,
                                           int lane, int maxrow) {
  int r = row0 + (lane & 15);
  if (r > maxrow) r = maxrow;
  const int kg = (lane >> 4) << 4;
  const _Float16* p = A + (long)r * lda + k0 + kg;
  HF16 u;
#pragma unroll
  for (int i = 0; i < 8; ++i) u.h[i] = p[i];
#pragma unroll
  for (int i = 0; i < 8; ++i) u.h[8 + i] = p[8 + i];
  return u.v;
}

// B fragment, source stored as N x K (B^T, like weight matrices W[n][k]).
__device__ __forceinline__ v16h frag_b_nk(const _Float16* __restrict__ B,
                                          int col0, int ldb, int k0,
                                          int lane, int maxn) {
  int n = col0 + (lane & 15);
  if (n > maxn) n = maxn;
  const int kb = k0 + ((lane >> 4) << 4);
  const _Float16* p = B + (long)n * ldb + kb;
  HF16 u;
#pragma unroll
  for (int i = 0; i < 16; ++i) u.h[i] = p[i];
  return u.v;
}

// B fragment, source stored as K x N row-major.
__device__ __forceinline__ v16h frag_b_kn(const _Float16* __restrict__ B,
                                          int col0, int ldb, int k0,
                                          int lane, int maxn) {
  int n = col0 + (lane & 15);
  if (n > maxn) n = maxn;
  const int kb = k0 + ((lane >> 4) << 4);
  HF16 u;
#pragma unroll
  for (int i = 0; i < 16; ++i) u.h[i] = B[(long)(kb + i) * ldb + n];
  return u.v;
}

#ifdef HAVE_TDM
// TDM: DMA `ndw` dwords from global `gsrc` into LDS offset `lds_off`.
// D# per CDNA5 ISA ch.8: group0 {count=1 | lds_addr | global_addr | type=2},
// group1 {data_size=4B, 1-D tensor: dim0=ndw, tile0=ndw, tile1=1}.
// This toolchain exposes the 6-arg builtin: (v4u, v8i, v4i, v4i, v8i, i32).
__device__ __forceinline__ void tdm_load_tile(const void* gsrc, unsigned lds_off,
                                              unsigned ndw) {
  unsigned long long ga = (unsigned long long)(uintptr_t)gsrc;
  v4u g0;
  g0[0] = 1u;                                    // count=1 (valid), user mode
  g0[1] = lds_off;                               // lds_addr (bytes)
  g0[2] = (unsigned)(ga & 0xFFFFFFFFu);          // global_addr[31:0]
  g0[3] = (unsigned)((ga >> 32) & 0x1FFFFFFu) | (2u << 30);  // [56:32] | type=2
  v8i g1;
  g1[0] = (int)(2u << 16);                       // wg_mask=0, data_size=2 (4B)
  g1[1] = (int)((ndw & 0xFFFFu) << 16);          // tensor_dim0[15:0] -> [31:16]
  g1[2] = (int)((ndw >> 16) | (1u << 16));       // tensor_dim0[31:16]; tensor_dim1=1
  g1[3] = (int)((ndw & 0xFFFFu) << 16);          // tensor_dim1 hi=0; tile_dim0=ndw
  g1[4] = 1;                                     // tile_dim1=1, tile_dim2=0
  g1[5] = (int)ndw;                              // tensor_dim0_stride[31:0]
  g1[6] = 0;
  g1[7] = 0;
  v4i g2 = {0, 0, 0, 0};
  v4i g3 = {0, 0, 0, 0};
  v8i g4 = {0, 0, 0, 0, 0, 0, 0, 0};
  __builtin_amdgcn_tensor_load_to_lds(g0, g1, g2, g3, g4, 0);
}
#endif

// ---------------------------------------------------------------------
// Generic batched WMMA GEMM: C[M,N] = A[M,K] * op(B) + bias
// BKN==0: B is N x K (B^T, weights). BKN==1: B is K x N row-major.
// F16OUT==1: store f16 with time-major row remap (b*T+t) -> (t*BB+b).
// K multiple of 32 (buffers zero-padded). 256 thr = 8 waves, tile 32x64.
// ---------------------------------------------------------------------
template <int BKN, int F16OUT>
__global__ void gemm_wmma_kernel(const _Float16* __restrict__ A,
                                 const _Float16* __restrict__ B,
                                 void* __restrict__ Cout,
                                 const float* __restrict__ bias,
                                 int M, int N, int K,
                                 int lda, int ldb, int ldc,
                                 long sA, long sB, long sC, int remapT) {
  const int batch = blockIdx.z;
  A += (long)batch * sA;
  B += (long)batch * sB;
  const int lane = threadIdx.x & 31;
  const int wid  = threadIdx.x >> 5;
  const int row0 = blockIdx.y * 32 + (wid >> 2) * 16;
  const int col0 = blockIdx.x * 64 + (wid & 3) * 16;

  v8f acc = {};
  for (int k0 = 0; k0 < K; k0 += 32) {
    if (k0 + 32 < K) {  // next A chunk -> global_prefetch_b8
      int rp = row0 + (lane & 15); if (rp > M - 1) rp = M - 1;
      __builtin_prefetch((const void*)(A + (long)rp * lda + k0 + 32), 0, 3);
    }
    v16h af = frag_a_f16(A, row0, lda, k0, lane, M - 1);
    v16h bf = BKN ? frag_b_kn(B, col0, ldb, k0, lane, N - 1)
                  : frag_b_nk(B, col0, ldb, k0, lane, N - 1);
    acc = __builtin_amdgcn_wmma_f32_16x16x32_f16(
        false, af, false, bf, (short)0, acc, false, false);
  }
#pragma unroll
  for (int e = 0; e < 8; ++e) {
    int r = row0 + e + ((lane >> 4) << 3);
    int c = col0 + (lane & 15);
    if (r < M && c < N) {
      float v = acc[e];
      if (bias) v += bias[c];
      if (F16OUT) {
        long ro = (long)(r % remapT) * BB + (r / remapT);  // time-major
        ((_Float16*)Cout)[(long)batch * sC + ro * ldc + c] = (_Float16)v;
      } else {
        ((float*)Cout)[(long)batch * sC + (long)r * ldc + c] = v;
      }
    }
  }
}

// ---------------------------------------------------------------------
// Persistent BiLSTM scan. grid.x = 2 (fwd, bwd); 896 threads = 28 waves.
// Per step: G(64x448) = xw16[t] + h(64x128)@Whh^T, 4x28 WMMA tiles; wave
// w: mtile = w&3, n-tiles 4*(w>>2)..+3. A-fragments from LDS (cached per
// step), B-fragments re-streamed from Whh^T (f16, WGP$/L2-resident) every
// step -- an empty asm makes the Wt pointer loop-variant so LICM cannot
// hoist 128 VGPRs of invariant fragments and spill them to scratch.
// xw16 is time-major f16; tile t+1 DMA'd by the TDM while step t computes.
// The G spill region aliases the retired xw tile. Gates in VALU.
// ---------------------------------------------------------------------
__global__ void __launch_bounds__(SCANT)
lstm_scan_kernel(const _Float16* __restrict__ xw_f,
                 const _Float16* __restrict__ xw_b,
                 const _Float16* __restrict__ Wt_f,
                 const _Float16* __restrict__ Wt_b,
                 float* __restrict__ out,  // [B][T][200]
                 int T) {
  const int rev = blockIdx.x;
  const _Float16* xw = rev ? xw_b : xw_f;       // [T][64][448] f16
  const _Float16* Wt = rev ? Wt_b : Wt_f;       // [448][128] f16
  const int outoff = rev ? 100 : 0;

  extern __shared__ char smem[];                // 16K h + 2x56K xw/G tiles
  _Float16* hbuf = (_Float16*)smem;             // [64][128]
  _Float16* xwb0 = (_Float16*)(smem + 16384);
  _Float16* xwb1 = (_Float16*)(smem + 16384 + 57344);

  const int tid  = threadIdx.x;
  const int lane = tid & 31;
  const int wid  = tid >> 5;                    // 28 waves
  const int mt   = wid & 3;
  const int ng0  = (wid >> 2) * 4;              // 4 consecutive n-tiles
  const int row0 = mt * 16;

  for (int i = tid; i < 64 * KHP; i += SCANT) hbuf[i] = (_Float16)0.0f;

  // stage xw tile for step 0
  const int t0 = rev ? (T - 1) : 0;
#ifdef HAVE_TDM
  if (wid == 0) {
    tdm_load_tile(xw + (long)t0 * (64 * NG),
                  (unsigned)(uintptr_t)xwb0, (64 * NG * 2) / 4);
    __builtin_amdgcn_s_wait_tensorcnt((short)0);
  }
  __syncthreads();
#else
  {
    const int* src = (const int*)(xw + (long)t0 * (64 * NG));
    int* dst = (int*)xwb0;
    for (int i = tid; i < (64 * NG) / 2; i += SCANT) dst[i] = src[i];
  }
  __syncthreads();
#endif

  float creg[8], hreg[8];
#pragma unroll
  for (int i = 0; i < 8; ++i) { creg[i] = 0.0f; hreg[i] = 0.0f; }

  unsigned long long wta = (unsigned long long)(uintptr_t)Wt;

  for (int step = 0; step < T; ++step) {
    const int t = rev ? (T - 1 - step) : step;
    _Float16* cur = (step & 1) ? xwb1 : xwb0;
    _Float16* nxt = (step & 1) ? xwb0 : xwb1;

    // Opaque update: keeps the Whh^T base loop-variant so its fragment
    // loads stay inside the step loop (stream from cache, not scratch).
    asm volatile("" : "+v"(wta));
    const _Float16* Wtp = (const _Float16*)(uintptr_t)wta;

    // ---- init accumulators from xw tile (LDS, f16) ----
    v8f acc[4];
#pragma unroll
    for (int nt = 0; nt < 4; ++nt) {
      const int col0 = (ng0 + nt) * 16;
      v8f a0;
#pragma unroll
      for (int e = 0; e < 8; ++e) {
        int b = row0 + e + ((lane >> 4) << 3);
        int c = col0 + (lane & 15);
        a0[e] = (float)cur[b * NG + c];
      }
      acc[nt] = a0;
    }
    __syncthreads();   // cur reads done; prior h writeback visible

#ifdef HAVE_TDM
    if (wid == 0 && step + 1 < T) {            // DMA next tile while computing
      const int tn = rev ? (T - 2 - step) : (step + 1);
      tdm_load_tile(xw + (long)tn * (64 * NG),
                    (unsigned)(uintptr_t)nxt, (64 * NG * 2) / 4);
    }
#endif

    // ---- GEMM: cache 4 A-fragments (LDS), stream B-fragments (global) ----
    v16h af[4];
#pragma unroll
    for (int k = 0; k < 4; ++k)
      af[k] = frag_a_f16(hbuf, row0, KHP, k * 32, lane, 63);
#pragma unroll
    for (int nt = 0; nt < 4; ++nt) {
      const int col0 = (ng0 + nt) * 16;
#pragma unroll
      for (int k = 0; k < 4; ++k) {
        v16h bf = frag_b_nk(Wtp, col0, KHP, k * 32, lane, NG - 1);
        acc[nt] = __builtin_amdgcn_wmma_f32_16x16x32_f16(
            false, af[k], false, bf, (short)0, acc[nt], false, false);
      }
    }
    __syncthreads();   // all hbuf / (aliased) reads done
    // ---- spill G (f16) into the retired xw tile region ----
#pragma unroll
    for (int nt = 0; nt < 4; ++nt) {
      const int col0 = (ng0 + nt) * 16;
#pragma unroll
      for (int e = 0; e < 8; ++e) {
        int b = row0 + e + ((lane >> 4) << 3);
        int c = col0 + (lane & 15);
        cur[b * NG + c] = (_Float16)acc[nt][e];
      }
    }
    __syncthreads();
    // ---- gates: 6400 (b,j) cells; c,h in registers ----
#pragma unroll
    for (int pi = 0; pi < 8; ++pi) {
      int idx = tid + pi * SCANT;
      if (idx < 6400) {
        int b = idx / 100, j = idx - b * 100;
        float gi = (float)cur[b * NG + j];
        float gf = (float)cur[b * NG + 112 + j];
        float gg = (float)cur[b * NG + 224 + j];
        float go = (float)cur[b * NG + 336 + j];
        float c  = sigf(gf) * creg[pi] + sigf(gi) * tanhf(gg);
        creg[pi] = c;
        float h  = sigf(go) * tanhf(c);
        hreg[pi] = h;
        out[((long)b * T + t) * 200 + outoff + j] = h;
      }
    }
    __syncthreads();   // gate reads of cur done
#pragma unroll
    for (int pi = 0; pi < 8; ++pi) {
      int idx = tid + pi * SCANT;
      if (idx < 6400) {
        int b = idx / 100, j = idx - b * 100;
        hbuf[b * KHP + j] = (_Float16)hreg[pi];
      }
    }
#ifdef HAVE_TDM
    if (wid == 0 && step + 1 < T) __builtin_amdgcn_s_wait_tensorcnt((short)0);
#else
    if (step + 1 < T) {  // synchronous restage of next tile
      const int tn = rev ? (T - 2 - step) : (step + 1);
      const int* src = (const int*)(xw + (long)tn * (64 * NG));
      int* dst = (int*)nxt;
      for (int i = tid; i < (64 * NG) / 2; i += SCANT) dst[i] = src[i];
    }
#endif
    __syncthreads();   // nxt tile ready, h writeback visible
  }
}

// ---------------------------------------------------------------------
// Char-CNN + glove gather: X0[word][0:100]=glove, [100:200]=char conv/max.
// ---------------------------------------------------------------------
__global__ void char_embed_kernel(const int* __restrict__ tok,
                                  const int* __restrict__ chars,   // [words][16]
                                  const float* __restrict__ glove, // [V][100]
                                  const float* __restrict__ cemb,  // [CV][8]
                                  const float* __restrict__ ck,    // [100][8][5]
                                  const float* __restrict__ cb,    // [100]
                                  float* __restrict__ X0, int nwords) {
  int word = blockIdx.x;
  if (word >= nwords) return;
  __shared__ float eS[8][16];
  int tid = threadIdx.x;
  if (tid < 16) {
    int cid = chars[(long)word * 16 + tid];
    for (int ce = 0; ce < 8; ++ce) eS[ce][tid] = cemb[cid * 8 + ce];
  }
  __syncthreads();
  if (tid < 100) {
    int tk = tok[word];
    X0[(long)word * 200 + tid] = glove[(long)tk * 100 + tid];
    float kf[40];
#pragma unroll
    for (int i = 0; i < 40; ++i) kf[i] = ck[tid * 40 + i];
    float m = -1e30f;
    for (int p = 0; p < 12; ++p) {
      float s = 0.f;
#pragma unroll
      for (int ce = 0; ce < 8; ++ce)
#pragma unroll
        for (int kw = 0; kw < 5; ++kw) s += eS[ce][p + kw] * kf[ce * 5 + kw];
      m = fmaxf(m, s);
    }
    float o = m + cb[tid];
    X0[(long)word * 200 + 100 + tid] = o > 0.f ? o : 0.f;
  }
}

// ---------------------------------------------------------------------
// small elementwise / pack kernels
// ---------------------------------------------------------------------
__global__ void f32_to_f16_pad_kernel(const float* __restrict__ src,
                                      _Float16* __restrict__ dst,
                                      int M, int K, int Kp,
                                      const float* __restrict__ scale) {
  long i = (long)blockIdx.x * blockDim.x + threadIdx.x;
  long total = (long)M * Kp;
  if (i >= total) return;
  int k = (int)(i % Kp);
  long m = i / Kp;
  float v = 0.f;
  if (k < K) { v = src[m * K + k]; if (scale) v *= scale[k]; }
  dst[i] = (_Float16)v;
}

__global__ void highway_combine_kernel(float* __restrict__ x,
                                       const float* __restrict__ F,
                                       const float* __restrict__ Gt, long n) {
  long i = (long)blockIdx.x * blockDim.x + threadIdx.x;
  if (i >= n) return;
  float g = sigf(Gt[i]);
  float f = fmaxf(F[i], 0.f);
  x[i] = g * f + (1.f - g) * x[i];
}

__global__ void pack_wih_kernel(const float* __restrict__ W,  // [400][din]
                                _Float16* __restrict__ dst,   // [448][dinp]
                                int din, int dinp) {
  long i = (long)blockIdx.x * blockDim.x + threadIdx.x;
  long total = (long)NG * dinp;
  if (i >= total) return;
  int k = (int)(i % dinp);
  int n = (int)(i / dinp);
  int gi = n / 112, j = n - gi * 112;
  float v = 0.f;
  if (j < 100 && k < din) v = W[(long)(gi * 100 + j) * din + k];
  dst[i] = (_Float16)v;
}

__global__ void pack_whh_kernel(const float* __restrict__ W,  // [400][100]
                                _Float16* __restrict__ dst) { // [448][128]
  int i = blockIdx.x * blockDim.x + threadIdx.x;
  if (i >= NG * KHP) return;
  int k = i % KHP, n = i / KHP;
  int gi = n / 112, j = n - gi * 112;
  float v = 0.f;
  if (j < 100 && k < 100) v = W[(gi * 100 + j) * 100 + k];
  dst[i] = (_Float16)v;
}

__global__ void pack_bias_kernel(const float* __restrict__ bih,
                                 const float* __restrict__ bhh,
                                 float* __restrict__ dst) {
  int n = blockIdx.x * blockDim.x + threadIdx.x;
  if (n >= NG) return;
  int gi = n / 112, j = n - gi * 112;
  dst[n] = (j < 100) ? bih[gi * 100 + j] + bhh[gi * 100 + j] : 0.f;
}

__global__ void rowdot_kernel(const float* __restrict__ A,
                              const float* __restrict__ w,
                              float* __restrict__ out, int rows, int K, int woff) {
  int r = blockIdx.x * blockDim.x + threadIdx.x;
  if (r >= rows) return;
  float s = 0.f;
  for (int k = 0; k < K; ++k) s += A[(long)r * K + k] * w[woff + k];
  out[r] = s;
}

// softmax over j of S[b,i,j] = SC + Ccw[b,i] + Qwq[b,j]; emits a (f16) and row max
__global__ void attn_softmax_kernel(const float* __restrict__ SC,
                                    const float* __restrict__ Ccw,
                                    const float* __restrict__ Qwq,
                                    _Float16* __restrict__ a16,
                                    float* __restrict__ Smax) {
  int bi = blockIdx.x * blockDim.x + threadIdx.x;
  if (bi >= BB * TCN) return;
  int b = bi >> 9;
  float row[TQW];
  float mx = -1e30f;
  float cterm = Ccw[bi];
#pragma unroll 8
  for (int j = 0; j < TQW; ++j) {
    float s = SC[(long)bi * TQW + j] + cterm + Qwq[b * TQW + j];
    row[j] = s;
    mx = fmaxf(mx, s);
  }
  Smax[bi] = mx;
  float ss = 0.f;
#pragma unroll 8
  for (int j = 0; j < TQW; ++j) { float e = __expf(row[j] - mx); row[j] = e; ss += e; }
  float inv = 1.f / ss;
#pragma unroll 8
  for (int j = 0; j < TQW; ++j) a16[(long)bi * TQW + j] = (_Float16)(row[j] * inv);
}

// bsm = softmax_i(Smax[b,:]); q2c[b,d] = sum_i bsm[i] * C[b,i,d]
__global__ void q2c_kernel(const float* __restrict__ Smax,
                           const float* __restrict__ C,
                           float* __restrict__ q2c) {
  int b = blockIdx.x, tid = threadIdx.x;
  __shared__ float red[256];
  float lm = -1e30f;
  for (int i = tid; i < TCN; i += 256) lm = fmaxf(lm, Smax[b * TCN + i]);
  red[tid] = lm; __syncthreads();
  for (int s = 128; s > 0; s >>= 1) { if (tid < s) red[tid] = fmaxf(red[tid], red[tid + s]); __syncthreads(); }
  float mx = red[0]; __syncthreads();
  float ls = 0.f;
  for (int i = tid; i < TCN; i += 256) ls += __expf(Smax[b * TCN + i] - mx);
  red[tid] = ls; __syncthreads();
  for (int s = 128; s > 0; s >>= 1) { if (tid < s) red[tid] += red[tid + s]; __syncthreads(); }
  float inv = 1.f / red[0];
  for (int d = tid; d < 200; d += 256) {
    float acc = 0.f;
    for (int i = 0; i < TCN; ++i)
      acc += __expf(Smax[b * TCN + i] - mx) * C[((long)b * TCN + i) * 200 + d];
    q2c[b * 200 + d] = acc * inv;
  }
}

// G = [C, c2q, C*c2q, C*q2c] emitted directly as f16 (K=800, 32-aligned)
__global__ void build_g_kernel(const float* __restrict__ C,
                               const float* __restrict__ c2q,
                               const float* __restrict__ q2c,
                               _Float16* __restrict__ Gh) {
  long i = (long)blockIdx.x * blockDim.x + threadIdx.x;
  const long total = (long)BB * TCN * 200;
  if (i >= total) return;
  long row = i / 200;
  int d = (int)(i - row * 200);
  int b = (int)(row >> 9);
  float c = C[i], cq = c2q[i], qc = q2c[b * 200 + d];
  long g0 = row * 800;
  Gh[g0 + d]       = (_Float16)c;
  Gh[g0 + 200 + d] = (_Float16)cq;
  Gh[g0 + 400 + d] = (_Float16)(c * cq);
  Gh[g0 + 600 + d] = (_Float16)(c * qc);
}

__global__ void logits_kernel(const _Float16* __restrict__ Gh,
                              const float* __restrict__ Mx,
                              const float* __restrict__ w,
                              float* __restrict__ out, int rows) {
  int r = blockIdx.x * blockDim.x + threadIdx.x;
  if (r >= rows) return;
  float s = 0.f;
  for (int k = 0; k < 800; ++k) s += (float)Gh[(long)r * 800 + k] * w[k];
  for (int k = 0; k < 200; ++k) s += Mx[(long)r * 200 + k] * w[800 + k];
  out[r] = s;
}

// ---------------------------------------------------------------------
// host-side helpers
// ---------------------------------------------------------------------
static void run_gemm(const _Float16* A, const _Float16* B, void* C, const float* bias,
                     int M, int N, int K, int lda, int ldb, int ldc,
                     long sA, long sB, long sC, int batches, int b_is_kn,
                     int f16out, int remapT, hipStream_t s) {
  dim3 grid((N + 63) / 64, (M + 31) / 32, batches);
  if (f16out)
    gemm_wmma_kernel<0, 1><<<grid, 256, 0, s>>>(A, B, C, bias, M, N, K, lda, ldb, ldc, sA, sB, sC, remapT);
  else if (b_is_kn)
    gemm_wmma_kernel<1, 0><<<grid, 256, 0, s>>>(A, B, C, bias, M, N, K, lda, ldb, ldc, sA, sB, sC, 1);
  else
    gemm_wmma_kernel<0, 0><<<grid, 256, 0, s>>>(A, B, C, bias, M, N, K, lda, ldb, ldc, sA, sB, sC, 1);
}

struct LstmScratch {
  _Float16 *Wihp_f, *Wihp_b, *Whhp_f, *Whhp_b;
  float *bs_f, *bs_b;
  _Float16 *xw_f, *xw_b;   // time-major f16 [T][64][448]
};

// param block layout (jax tree flatten, sorted keys): bwd{Whh,Wih,bhh,bih}, fwd{...}
static void run_bilstm(void* const* d_in, int pbase,
                       const _Float16* Xh, int Mrows, int din, int dinp, int T,
                       LstmScratch& ws, float* out, hipStream_t s) {
  const float* Whh_b = (const float*)d_in[pbase + 0];
  const float* Wih_b = (const float*)d_in[pbase + 1];
  const float* bhh_b = (const float*)d_in[pbase + 2];
  const float* bih_b = (const float*)d_in[pbase + 3];
  const float* Whh_f = (const float*)d_in[pbase + 4];
  const float* Wih_f = (const float*)d_in[pbase + 5];
  const float* bhh_f = (const float*)d_in[pbase + 6];
  const float* bih_f = (const float*)d_in[pbase + 7];
  long nwih = (long)NG * dinp;
  int g1 = (int)((nwih + 255) / 256);
  pack_wih_kernel<<<g1, 256, 0, s>>>(Wih_f, ws.Wihp_f, din, dinp);
  pack_wih_kernel<<<g1, 256, 0, s>>>(Wih_b, ws.Wihp_b, din, dinp);
  pack_whh_kernel<<<(NG * KHP + 255) / 256, 256, 0, s>>>(Whh_f, ws.Whhp_f);
  pack_whh_kernel<<<(NG * KHP + 255) / 256, 256, 0, s>>>(Whh_b, ws.Whhp_b);
  pack_bias_kernel<<<2, 256, 0, s>>>(bih_f, bhh_f, ws.bs_f);
  pack_bias_kernel<<<2, 256, 0, s>>>(bih_b, bhh_b, ws.bs_b);
  // xW for all timesteps -> time-major f16 tiles for the scan
  run_gemm(Xh, ws.Wihp_f, ws.xw_f, ws.bs_f, Mrows, NG, dinp, dinp, dinp, NG,
           0, 0, 0, 1, 0, 1, T, s);
  run_gemm(Xh, ws.Wihp_b, ws.xw_b, ws.bs_b, Mrows, NG, dinp, dinp, dinp, NG,
           0, 0, 0, 1, 0, 1, T, s);
  lstm_scan_kernel<<<2, SCANT, 16384 + 2 * 57344, s>>>(
      ws.xw_f, ws.xw_b, ws.Whhp_f, ws.Whhp_b, out, T);
}

static void run_highway(float* X0, _Float16* xh, float* F, float* Gt,
                        const _Float16* fW16, const _Float16* gW16,
                        const float* fb, const float* gb, int Mrows, hipStream_t s) {
  for (int l = 0; l < 2; ++l) {
    long tot = (long)Mrows * 224;
    f32_to_f16_pad_kernel<<<(int)((tot + 255) / 256), 256, 0, s>>>(X0, xh, Mrows, 200, 224, nullptr);
    run_gemm(xh, fW16 + (size_t)l * 200 * 224, F, fb + l * 200, Mrows, 200, 224, 224, 224, 200, 0, 0, 0, 1, 0, 0, 0, s);
    run_gemm(xh, gW16 + (size_t)l * 200 * 224, Gt, gb + l * 200, Mrows, 200, 224, 224, 224, 200, 0, 0, 0, 1, 0, 0, 0, s);
    long n = (long)Mrows * 200;
    highway_combine_kernel<<<(int)((n + 255) / 256), 256, 0, s>>>(X0, F, Gt, n);
  }
  long tot = (long)Mrows * 224;
  f32_to_f16_pad_kernel<<<(int)((tot + 255) / 256), 256, 0, s>>>(X0, xh, Mrows, 200, 224, nullptr);
}

// input indices (setup_inputs order; params jax-tree-flattened with sorted keys)
enum {
  IN_CTX = 0, IN_QUES, IN_CHAR_CTX, IN_CHAR_QUES,
  IN_CHAR_EMB, IN_CONV_B, IN_CONV_K,
  IN_CTXL = 7,                  // ctx_lstm: bwd Whh,Wih,bhh,bih; fwd ...
  IN_ENDL = 15,                 // end
  IN_GLOVE = 23,
  IN_HW_FW = 24, IN_HW_FB, IN_HW_GW, IN_HW_GB,
  IN_M1L = 28, IN_M2L = 36,
  IN_SIM_W = 44, IN_W_END = 45, IN_W_START = 46
};

extern "C" void kernel_launch(void* const* d_in, const int* in_sizes, int n_in,
                              void* d_out, int out_size, void* d_ws, size_t ws_size,
                              hipStream_t stream) {
  (void)in_sizes; (void)n_in; (void)out_size; (void)ws_size;
  const int* ctx       = (const int*)d_in[IN_CTX];
  const int* ques      = (const int*)d_in[IN_QUES];
  const int* char_ctx  = (const int*)d_in[IN_CHAR_CTX];
  const int* char_ques = (const int*)d_in[IN_CHAR_QUES];
  const float* cemb    = (const float*)d_in[IN_CHAR_EMB];
  const float* convb   = (const float*)d_in[IN_CONV_B];
  const float* convk   = (const float*)d_in[IN_CONV_K];
  const float* glove   = (const float*)d_in[IN_GLOVE];
  const float* hw_fW   = (const float*)d_in[IN_HW_FW];
  const float* hw_fb   = (const float*)d_in[IN_HW_FB];
  const float* hw_gW   = (const float*)d_in[IN_HW_GW];
  const float* hw_gb   = (const float*)d_in[IN_HW_GB];
  const float* sim_w   = (const float*)d_in[IN_SIM_W];
  const float* w_end   = (const float*)d_in[IN_W_END];
  const float* w_start = (const float*)d_in[IN_W_START];
  float* out = (float*)d_out;

  // ---- workspace bump allocator ----
  char* base = (char*)d_ws;
  size_t off = 0;
  auto alloc = [&](size_t bytes) -> void* {
    void* p = base + off;
    off += (bytes + 255) & ~(size_t)255;
    return p;
  };
  const int MC = BB * TCN, MQ = BB * TQW;
  float*    X0c  = (float*)alloc((size_t)MC * 200 * 4);
  float*    X0q  = (float*)alloc((size_t)MQ * 200 * 4);
  _Float16* xhc  = (_Float16*)alloc((size_t)MC * 224 * 2);
  _Float16* xhq  = (_Float16*)alloc((size_t)MQ * 224 * 2);
  float*    Fbuf = (float*)alloc((size_t)MC * 200 * 4);
  float*    Gtb  = (float*)alloc((size_t)MC * 200 * 4);
  _Float16* fW16 = (_Float16*)alloc((size_t)2 * 200 * 224 * 2);
  _Float16* gW16 = (_Float16*)alloc((size_t)2 * 200 * 224 * 2);
  LstmScratch ls;
  ls.Wihp_f = (_Float16*)alloc((size_t)NG * 800 * 2);
  ls.Wihp_b = (_Float16*)alloc((size_t)NG * 800 * 2);
  ls.Whhp_f = (_Float16*)alloc((size_t)NG * KHP * 2);
  ls.Whhp_b = (_Float16*)alloc((size_t)NG * KHP * 2);
  ls.bs_f   = (float*)alloc(NG * 4);
  ls.bs_b   = (float*)alloc(NG * 4);
  ls.xw_f   = (_Float16*)alloc((size_t)MC * NG * 2);
  ls.xw_b   = (_Float16*)alloc((size_t)MC * NG * 2);
  float*    Cc   = (float*)alloc((size_t)MC * 200 * 4);
  float*    Qq   = (float*)alloc((size_t)MQ * 200 * 4);
  _Float16* Cm16 = (_Float16*)alloc((size_t)MC * 224 * 2);
  _Float16* Qh16 = (_Float16*)alloc((size_t)MQ * 224 * 2);
  float*    Ccw  = (float*)alloc((size_t)MC * 4);
  float*    Qwq  = (float*)alloc((size_t)MQ * 4);
  float*    Sbuf = (float*)alloc((size_t)MC * TQW * 4);
  _Float16* a16  = (_Float16*)alloc((size_t)MC * TQW * 2);
  float*    Smax = (float*)alloc((size_t)MC * 4);
  float*    c2q  = (float*)alloc((size_t)MC * 200 * 4);
  float*    q2c  = (float*)alloc((size_t)BB * 200 * 4);
  _Float16* Gh   = (_Float16*)alloc((size_t)MC * 800 * 2);
  float* M1 = Fbuf;   // reuse (dead after highway)
  float* Mm = Gtb;    // reuse
  float* M2 = c2q;    // reuse (dead after build_g)

  // 1) embeddings + char CNN
  char_embed_kernel<<<MC, 128, 0, stream>>>(ctx, char_ctx, glove, cemb, convk, convb, X0c, MC);
  char_embed_kernel<<<MQ, 128, 0, stream>>>(ques, char_ques, glove, cemb, convk, convb, X0q, MQ);

  // 2) highway weights -> f16 (both layers at once: 400 x 200 -> 400 x 224)
  f32_to_f16_pad_kernel<<<(400 * 224 + 255) / 256, 256, 0, stream>>>(hw_fW, fW16, 400, 200, 224, nullptr);
  f32_to_f16_pad_kernel<<<(400 * 224 + 255) / 256, 256, 0, stream>>>(hw_gW, gW16, 400, 200, 224, nullptr);

  // 3) highway (ctx, ques) -> xhc/xhq f16
  run_highway(X0c, xhc, Fbuf, Gtb, fW16, gW16, hw_fb, hw_gb, MC, stream);
  run_highway(X0q, xhq, Fbuf, Gtb, fW16, gW16, hw_fb, hw_gb, MQ, stream);

  // 4) encoder BiLSTM (shared params) -> C, Q
  run_bilstm(d_in, IN_CTXL, xhc, MC, 200, 224, TCN, ls, Cc, stream);
  run_bilstm(d_in, IN_CTXL, xhq, MQ, 200, 224, TQW, ls, Qq, stream);

  // 5) attention
  rowdot_kernel<<<(MC + 255) / 256, 256, 0, stream>>>(Cc, sim_w, Ccw, MC, 200, 0);
  rowdot_kernel<<<(MQ + 255) / 256, 256, 0, stream>>>(Qq, sim_w, Qwq, MQ, 200, 200);
  f32_to_f16_pad_kernel<<<(int)(((long)MC * 224 + 255) / 256), 256, 0, stream>>>(Cc, Cm16, MC, 200, 224, sim_w + 400);
  f32_to_f16_pad_kernel<<<(int)(((long)MQ * 224 + 255) / 256), 256, 0, stream>>>(Qq, Qh16, MQ, 200, 224, nullptr);
  // SC[b] = (C*wm)[512x224] @ Q[64x224]^T  (batched WMMA)
  run_gemm(Cm16, Qh16, Sbuf, nullptr, TCN, TQW, 224, 224, 224, TQW,
           (long)TCN * 224, (long)TQW * 224, (long)TCN * TQW, BB, 0, 0, 0, stream);
  attn_softmax_kernel<<<(MC + 255) / 256, 256, 0, stream>>>(Sbuf, Ccw, Qwq, a16, Smax);
  // c2q[b] = a[512x64] @ Q[64x200]  (batched WMMA, B in KxN mode over Qh16)
  run_gemm(a16, Qh16, c2q, nullptr, TCN, 200, TQW, TQW, 224, 200,
           (long)TCN * TQW, (long)TQW * 224, (long)TCN * 200, BB, 1, 0, 0, stream);
  q2c_kernel<<<BB, 256, 0, stream>>>(Smax, Cc, q2c);
  build_g_kernel<<<(int)(((long)MC * 200 + 255) / 256), 256, 0, stream>>>(Cc, c2q, q2c, Gh);

  // 6) modeling stack: mod1 (din 800), mod2, end
  run_bilstm(d_in, IN_M1L, Gh, MC, 800, 800, TCN, ls, M1, stream);
  f32_to_f16_pad_kernel<<<(int)(((long)MC * 224 + 255) / 256), 256, 0, stream>>>(M1, xhc, MC, 200, 224, nullptr);
  run_bilstm(d_in, IN_M2L, xhc, MC, 200, 224, TCN, ls, Mm, stream);
  f32_to_f16_pad_kernel<<<(int)(((long)MC * 224 + 255) / 256), 256, 0, stream>>>(Mm, xhc, MC, 200, 224, nullptr);
  run_bilstm(d_in, IN_ENDL, xhc, MC, 200, 224, TCN, ls, M2, stream);

  // 7) logits: p1 = [G,M] @ w_start ; p2 = [G,M2] @ w_end
  logits_kernel<<<(MC + 255) / 256, 256, 0, stream>>>(Gh, Mm, w_start, out, MC);
  logits_kernel<<<(MC + 255) / 256, 256, 0, stream>>>(Gh, M2, w_end, out + MC, MC);
}
